// LCPFormerBlock_89077621719501
// MI455X (gfx1250) — compile-verified, matrix operand order checked by reading, not run on hardware
//
#include <hip/hip_runtime.h>
#include <cmath>

typedef __bf16 bf16;
typedef __attribute__((ext_vector_type(16))) __bf16 v16bf;
typedef __attribute__((ext_vector_type(8)))  float  v8f;

constexpr int kB  = 4;
constexpr int kN  = 8192;
constexpr int kC  = 128;
constexpr int kNP = 2048;
constexpr int kNS = 32;
constexpr int kOut = 256;
constexpr int kM  = kB * kNP * kNS;   // 262144 tokens

// ---------------------------------------------------------------------------
// helpers
// ---------------------------------------------------------------------------
__device__ inline v16bf ld_frag(const bf16* base) {
  v16bf r;
  uint4* p = reinterpret_cast<uint4*>(&r);
  p[0] = *reinterpret_cast<const uint4*>(base);
  p[1] = *reinterpret_cast<const uint4*>(base + 16);
  return r;
}

// CDNA5 async global -> LDS copy (16B per lane, tracked by ASYNCcnt)
__device__ inline void async_cp_b128(uint32_t lds_addr, const void* gaddr) {
  asm volatile("global_load_async_to_lds_b128 %0, %1, off"
               :: "v"(lds_addr), "v"(gaddr) : "memory");
}
__device__ inline void wait_async_le4() {
  asm volatile("s_wait_asynccnt 0x4" ::: "memory");
}
__device__ inline void wait_async_0() {
  asm volatile("s_wait_asynccnt 0x0" ::: "memory");
}

// ---------------------------------------------------------------------------
// f32 -> bf16 conversion
// ---------------------------------------------------------------------------
__global__ void f2bf_kernel(const float* __restrict__ s, bf16* __restrict__ d, int n) {
  int i = blockIdx.x * 256 + threadIdx.x;
  if (i < n) d[i] = (bf16)s[i];
}

// ---------------------------------------------------------------------------
// Farthest point sampling: one block per batch, 1024 threads, 8 pts/thread
// ---------------------------------------------------------------------------
__global__ void __launch_bounds__(1024) fps_kernel(const float* __restrict__ xyz,
                                                   int* __restrict__ inds,
                                                   float* __restrict__ newxyz) {
  const int b = blockIdx.x, tid = threadIdx.x;
  __shared__ float s_best[32];
  __shared__ int   s_bidx[32];
  __shared__ float s_last[3];
  float px[8], py[8], pz[8], dd[8];
#pragma unroll
  for (int j = 0; j < 8; ++j) {
    int p = tid + j * 1024;
    size_t base = ((size_t)b * kN + p) * 3;
    px[j] = xyz[base + 0]; py[j] = xyz[base + 1]; pz[j] = xyz[base + 2];
    dd[j] = 1e10f;
  }
  if (tid == 0) {
    inds[b * kNP] = 0;
    size_t sb = (size_t)b * kN * 3;
    for (int k = 0; k < 3; ++k) {
      float v = xyz[sb + k];
      newxyz[(size_t)b * kNP * 3 + k] = v;
      s_last[k] = v;
    }
  }
  __syncthreads();
  for (int it = 1; it < kNP; ++it) {
    float lx = s_last[0], ly = s_last[1], lz = s_last[2];
    float bd = -1.0f; int bi = 0;
#pragma unroll
    for (int j = 0; j < 8; ++j) {
      float dx = px[j] - lx, dy = py[j] - ly, dz = pz[j] - lz;
      float d2 = dx * dx + dy * dy + dz * dz;
      dd[j] = fminf(dd[j], d2);
      if (dd[j] > bd) { bd = dd[j]; bi = tid + j * 1024; }
    }
    for (int m = 16; m >= 1; m >>= 1) {
      float od = __shfl_xor(bd, m, 32);
      int   oi = __shfl_xor(bi, m, 32);
      if (od > bd) { bd = od; bi = oi; }
    }
    if ((tid & 31) == 0) { s_best[tid >> 5] = bd; s_bidx[tid >> 5] = bi; }
    __syncthreads();
    if (tid == 0) {
      float best = -1.0f; int besti = 0;
      for (int w = 0; w < 32; ++w)
        if (s_best[w] > best) { best = s_best[w]; besti = s_bidx[w]; }
      inds[b * kNP + it] = besti;
      size_t sb = ((size_t)b * kN + besti) * 3;
      for (int k = 0; k < 3; ++k) {
        float v = xyz[sb + k];
        newxyz[((size_t)b * kNP + it) * 3 + k] = v;
        s_last[k] = v;
      }
    }
    __syncthreads();
  }
}

// ---------------------------------------------------------------------------
// KNN top-32: one thread per query, LDS-tiled scan, register insertion sort
// ---------------------------------------------------------------------------
__global__ void __launch_bounds__(256) knn_kernel(const float* __restrict__ xyz,
                                                  const float* __restrict__ newxyz,
                                                  int* __restrict__ knn) {
  __shared__ float sx[256 * 3];
  const int q = blockIdx.x * 256 + threadIdx.x;   // blocks never straddle batches
  const int b = q / kNP;
  const float qx = newxyz[(size_t)q * 3 + 0];
  const float qy = newxyz[(size_t)q * 3 + 1];
  const float qz = newxyz[(size_t)q * 3 + 2];
  float bd[kNS]; int bi[kNS];
#pragma unroll
  for (int u = 0; u < kNS; ++u) { bd[u] = 1e30f; bi[u] = 0; }
  for (int t0 = 0; t0 < kN; t0 += 256) {
#pragma unroll
    for (int k = 0; k < 3; ++k)
      sx[threadIdx.x * 3 + k] = xyz[((size_t)b * kN + t0 + threadIdx.x) * 3 + k];
    __syncthreads();
    for (int j = 0; j < 256; ++j) {
      float dx = sx[j * 3 + 0] - qx;
      float dy = sx[j * 3 + 1] - qy;
      float dz = sx[j * 3 + 2] - qz;
      float d2 = dx * dx + dy * dy + dz * dz;
      if (d2 < bd[kNS - 1]) {
        bd[kNS - 1] = d2; bi[kNS - 1] = t0 + j;
#pragma unroll
        for (int u = kNS - 1; u > 0; --u) {
          if (bd[u] < bd[u - 1]) {
            float td = bd[u]; bd[u] = bd[u - 1]; bd[u - 1] = td;
            int   ti = bi[u]; bi[u] = bi[u - 1]; bi[u - 1] = ti;
          }
        }
      }
    }
    __syncthreads();
  }
#pragma unroll
  for (int u = 0; u < kNS; ++u) knn[(size_t)q * kNS + u] = bi[u];
}

// ---------------------------------------------------------------------------
// Token build: X[m][c] = feat_gather + positional_encoding  (bf16 out)
// ---------------------------------------------------------------------------
__global__ void __launch_bounds__(128) build_tokens_kernel(
    const float* __restrict__ feat,     // (B, C, N)
    const float* __restrict__ xyz, const float* __restrict__ newxyz,
    const int* __restrict__ knn,
    const float* __restrict__ w1, const float* __restrict__ bn_g,
    const float* __restrict__ bn_b, const float* __restrict__ w2,
    const float* __restrict__ b2, bf16* __restrict__ X) {
  __shared__ float sh[32 * 64];
  __shared__ float sloc[32 * 3];
  const int win = blockIdx.x;
  const int b = win / kNP, p = win % kNP;
  const int t = threadIdx.x;
  if (t < 96) {
    int s = t / 3, k = t % 3;
    int id = knn[(size_t)win * kNS + s];
    sloc[s * 3 + k] = (xyz[((size_t)b * kN + id) * 3 + k] -
                       newxyz[((size_t)b * kNP + p) * 3 + k]) * 2.5f; // 1/0.4
  }
  __syncthreads();
  const float bnr = rsqrtf(1.0f + 1e-5f);
#pragma unroll
  for (int jj = 0; jj < 16; ++jj) {
    int v = t + jj * 128;        // 0..2047
    int s = v >> 6, o = v & 63;
    float h = w1[o * 3 + 0] * sloc[s * 3 + 0] +
              w1[o * 3 + 1] * sloc[s * 3 + 1] +
              w1[o * 3 + 2] * sloc[s * 3 + 2];
    sh[s * 64 + o] = h * (bn_g[o] * bnr) + bn_b[o];
  }
  __syncthreads();
  const int c = t;
  const float bb = b2[c];
  for (int s = 0; s < 32; ++s) {
    float pos = bb;
    for (int o = 0; o < 64; ++o) pos += w2[c * 64 + o] * sh[s * 64 + o];
    int id = knn[(size_t)win * kNS + s];
    float fv = feat[((size_t)b * kC + c) * kN + id];
    X[((size_t)win * kNS + s) * kC + c] = (bf16)(fv + pos);
  }
}

// ---------------------------------------------------------------------------
// bf16 GEMM: out[m][n] = act(sum_k A[m][k]*W[n][k] * scale[n]*smul + bias[n])
// A: MxK row-major, W: NxK row-major (both bf16).
// Block tile 128x128, 8 waves, each wave 32x64 (2x4 WMMA tiles).
// K staged 32-wide into double-buffered LDS via async global->LDS DMA
// (GLOBAL_LOAD_ASYNC_TO_LDS_B128 + s_wait_asynccnt).
// ---------------------------------------------------------------------------
__global__ void __launch_bounds__(256) gemm_bf16_kernel(
    const bf16* __restrict__ A, const bf16* __restrict__ W,
    const float* __restrict__ scale, float smul,
    const float* __restrict__ bias, bf16* __restrict__ out,
    int M, int N, int K, int act) {
  __shared__ bf16 sA[2][128 * 32];
  __shared__ bf16 sB[2][128 * 32];
  const int tid = threadIdx.x;
  const int lane = tid & 31, wave = tid >> 5;
  const int wm = wave & 3;          // 4 waves along M, 32 rows each
  const int wn = wave >> 2;         // 2 waves along N, 64 cols each
  const int m0 = blockIdx.y * 128;
  const int n0 = blockIdx.x * 128;
  const int lr = lane & 15;
  const int kh = (lane >> 4) * 8;

  // per-thread staging coordinates: 4 x 16B per thread per chunk
  const int rowA0 = tid >> 2,           c8A0 = (tid & 3) * 8;
  const int rowA1 = (tid + 256) >> 2,   c8A1 = ((tid + 256) & 3) * 8;

  v8f acc[2][4];
#pragma unroll
  for (int i = 0; i < 2; ++i)
#pragma unroll
    for (int j = 0; j < 4; ++j) acc[i][j] = (v8f){0, 0, 0, 0, 0, 0, 0, 0};

  auto stage = [&](int buf, int k0) {
    async_cp_b128((uint32_t)(uintptr_t)&sA[buf][rowA0 * 32 + c8A0],
                  &A[(size_t)(m0 + rowA0) * K + k0 + c8A0]);
    async_cp_b128((uint32_t)(uintptr_t)&sA[buf][rowA1 * 32 + c8A1],
                  &A[(size_t)(m0 + rowA1) * K + k0 + c8A1]);
    async_cp_b128((uint32_t)(uintptr_t)&sB[buf][rowA0 * 32 + c8A0],
                  &W[(size_t)(n0 + rowA0) * K + k0 + c8A0]);
    async_cp_b128((uint32_t)(uintptr_t)&sB[buf][rowA1 * 32 + c8A1],
                  &W[(size_t)(n0 + rowA1) * K + k0 + c8A1]);
  };

  const int nk = K >> 5;
  stage(0, 0);
  for (int kc = 0; kc < nk; ++kc) {
    const int cur = kc & 1;
    if (kc + 1 < nk) {
      stage(cur ^ 1, (kc + 1) * 32);
      wait_async_le4();               // chunk kc landed in LDS (in-order)
    } else {
      wait_async_0();
    }
    __syncthreads();

    const bf16* bufA = sA[cur];
    const bf16* bufB = sB[cur];
    v16bf af[2], bw[4];
#pragma unroll
    for (int mi = 0; mi < 2; ++mi)
      af[mi] = ld_frag(&bufA[(wm * 32 + mi * 16 + lr) * 32 + kh]);
#pragma unroll
    for (int ni = 0; ni < 4; ++ni)
      bw[ni] = ld_frag(&bufB[(wn * 64 + ni * 16 + lr) * 32 + kh]);
#pragma unroll
    for (int mi = 0; mi < 2; ++mi)
#pragma unroll
      for (int ni = 0; ni < 4; ++ni)
        acc[mi][ni] = __builtin_amdgcn_wmma_f32_16x16x32_bf16(
            false, af[mi], false, bw[ni], (short)0, acc[mi][ni], false, false);
    __syncthreads();                  // protect buffer reuse (WAR vs next stage)
  }

  const int rhalf = (lane >> 4) ? 8 : 0;
#pragma unroll
  for (int ni = 0; ni < 4; ++ni) {
    int col = n0 + wn * 64 + ni * 16 + lr;
    float scl = smul * (scale ? scale[col] : 1.0f);
    float bia = bias ? bias[col] : 0.0f;
#pragma unroll
    for (int mi = 0; mi < 2; ++mi) {
      int rb = m0 + wm * 32 + mi * 16 + rhalf;
#pragma unroll
      for (int r = 0; r < 8; ++r) {
        float v = acc[mi][ni][r] * scl + bia;
        if (act) v = v > 0.0f ? v : 0.0f;
        out[(size_t)(rb + r) * N + col] = (bf16)v;
      }
    }
  }
}

// ---------------------------------------------------------------------------
// Attention: one wave per (window, head). Q,K,V slices of qkv (M x 384).
// S = QK^T via WMMA, softmax in-register, O = P*V^T via WMMA (LDS staging).
// ---------------------------------------------------------------------------
__global__ void __launch_bounds__(256) attn_kernel(const bf16* __restrict__ qkv,
                                                   bf16* __restrict__ outb) {
  __shared__ bf16 sP[8][1024];
  __shared__ bf16 sVt[8][1024];
  const int tid = threadIdx.x, lane = tid & 31, wave = tid >> 5;
  const int gw = blockIdx.x * 8 + wave;
  const int win = gw >> 2, head = gw & 3;
  const int rb = win * 32;
  const int lr = lane & 15, kh = (lane >> 4) * 8;
  const int qoff = head * 32, koff = 128 + head * 32, voff = 256 + head * 32;

  v16bf qf[2], kf[2];
#pragma unroll
  for (int mi = 0; mi < 2; ++mi)
    qf[mi] = ld_frag(&qkv[(size_t)(rb + mi * 16 + lr) * 384 + qoff + kh]);
#pragma unroll
  for (int ni = 0; ni < 2; ++ni)
    kf[ni] = ld_frag(&qkv[(size_t)(rb + ni * 16 + lr) * 384 + koff + kh]);

  v8f s[2][2];
#pragma unroll
  for (int i = 0; i < 2; ++i)
#pragma unroll
    for (int j = 0; j < 2; ++j) s[i][j] = (v8f){0, 0, 0, 0, 0, 0, 0, 0};
#pragma unroll
  for (int mi = 0; mi < 2; ++mi)
#pragma unroll
    for (int ni = 0; ni < 2; ++ni)
      s[mi][ni] = __builtin_amdgcn_wmma_f32_16x16x32_bf16(
          false, qf[mi], false, kf[ni], (short)0, s[mi][ni], false, false);

  // softmax over rows; each acc vgpr r holds two rows (one per 16-lane half)
  const float scl = 0.17677669529663687f;   // 1/sqrt(32)
  float pr[2][2][8];
#pragma unroll
  for (int mi = 0; mi < 2; ++mi) {
#pragma unroll
    for (int r = 0; r < 8; ++r) {
      float mx = fmaxf(s[mi][0][r], s[mi][1][r]);
      for (int m = 8; m >= 1; m >>= 1) mx = fmaxf(mx, __shfl_xor(mx, m, 32));
      float e0 = __expf((s[mi][0][r] - mx) * scl);
      float e1 = __expf((s[mi][1][r] - mx) * scl);
      float su = e0 + e1;
      for (int m = 8; m >= 1; m >>= 1) su += __shfl_xor(su, m, 32);
      float inv = 1.0f / su;
      pr[mi][0][r] = e0 * inv;
      pr[mi][1][r] = e1 * inv;
    }
  }
  const int rhalf = (lane >> 4) ? 8 : 0;
#pragma unroll
  for (int mi = 0; mi < 2; ++mi)
#pragma unroll
    for (int ni = 0; ni < 2; ++ni)
#pragma unroll
      for (int r = 0; r < 8; ++r) {
        int m = mi * 16 + r + rhalf;
        int n = ni * 16 + lr;
        sP[wave][m * 32 + n] = (bf16)pr[mi][ni][r];
      }
  {
    const bf16* vp = &qkv[(size_t)(rb + lane) * 384 + voff];
#pragma unroll
    for (int d = 0; d < 32; ++d) sVt[wave][d * 32 + lane] = vp[d];
  }
  __syncthreads();

  v16bf pf[2], vf[2];
#pragma unroll
  for (int mi = 0; mi < 2; ++mi)
    pf[mi] = ld_frag(&sP[wave][(mi * 16 + lr) * 32 + kh]);
#pragma unroll
  for (int ni = 0; ni < 2; ++ni)
    vf[ni] = ld_frag(&sVt[wave][(ni * 16 + lr) * 32 + kh]);

  v8f o[2][2];
#pragma unroll
  for (int i = 0; i < 2; ++i)
#pragma unroll
    for (int j = 0; j < 2; ++j) o[i][j] = (v8f){0, 0, 0, 0, 0, 0, 0, 0};
#pragma unroll
  for (int mi = 0; mi < 2; ++mi)
#pragma unroll
    for (int ni = 0; ni < 2; ++ni)
      o[mi][ni] = __builtin_amdgcn_wmma_f32_16x16x32_bf16(
          false, pf[mi], false, vf[ni], (short)0, o[mi][ni], false, false);

#pragma unroll
  for (int mi = 0; mi < 2; ++mi)
#pragma unroll
    for (int ni = 0; ni < 2; ++ni) {
      int col = head * 32 + ni * 16 + lr;
#pragma unroll
      for (int r = 0; r < 8; ++r) {
        int row = rb + mi * 16 + r + rhalf;
        outb[(size_t)row * kC + col] = (bf16)o[mi][ni][r];
      }
    }
}

// ---------------------------------------------------------------------------
// X = LayerNorm(X + O), one wave per row (d=128)
// ---------------------------------------------------------------------------
__global__ void __launch_bounds__(256) add_ln_kernel(bf16* __restrict__ X,
                                                     const bf16* __restrict__ O,
                                                     const float* __restrict__ g,
                                                     const float* __restrict__ bt) {
  const int lane = threadIdx.x & 31;
  const size_t row = (size_t)blockIdx.x * 8 + (threadIdx.x >> 5);
  float v[4], sum = 0.0f, sq = 0.0f;
#pragma unroll
  for (int j = 0; j < 4; ++j) {
    int c = lane + j * 32;
    v[j] = (float)X[row * kC + c] + (float)O[row * kC + c];
    sum += v[j]; sq += v[j] * v[j];
  }
  for (int m = 16; m >= 1; m >>= 1) {
    sum += __shfl_xor(sum, m, 32);
    sq  += __shfl_xor(sq, m, 32);
  }
  float mean = sum * (1.0f / kC);
  float var = sq * (1.0f / kC) - mean * mean;
  float inv = rsqrtf(var + 1e-5f);
#pragma unroll
  for (int j = 0; j < 4; ++j) {
    int c = lane + j * 32;
    X[row * kC + c] = (bf16)((v[j] - mean) * inv * g[c] + bt[c]);
  }
}

// ---------------------------------------------------------------------------
// scatter-mean (accumulate) / divide
// ---------------------------------------------------------------------------
__global__ void __launch_bounds__(128) scatter_kernel(const bf16* __restrict__ L,
                                                      const int* __restrict__ knn,
                                                      float* __restrict__ cross,
                                                      float* __restrict__ cnt) {
  const size_t m = blockIdx.x;
  const int c = threadIdx.x;
  const int b = (int)(m / ((size_t)kNP * kNS));
  const int n = knn[m];
  atomicAdd(&cross[((size_t)b * kC + c) * kN + n], (float)L[m * kC + c]);
  if (c == 0) atomicAdd(&cnt[(size_t)b * kN + n], 1.0f);
}

__global__ void divide_kernel(float* __restrict__ cross, const float* __restrict__ cnt) {
  size_t i = (size_t)blockIdx.x * 256 + threadIdx.x;
  if (i >= (size_t)kB * kC * kN) return;
  int n = (int)(i % kN);
  int b = (int)(i / ((size_t)kC * kN));
  cross[i] /= fmaxf(cnt[(size_t)b * kN + n], 1.0f);
}

// ---------------------------------------------------------------------------
// max-pool over window dim -> (B, OUT, NP)
// ---------------------------------------------------------------------------
__global__ void maxpool_kernel(const bf16* __restrict__ Y, float* __restrict__ outp) {
  size_t i = (size_t)blockIdx.x * 256 + threadIdx.x;
  if (i >= (size_t)kB * kOut * kNP) return;
  int p = (int)(i % kNP);
  size_t r = i / kNP;
  int o = (int)(r % kOut);
  int b = (int)(r / kOut);
  size_t base = (((size_t)b * kNP + p) * kNS) * kOut + o;
  float mx = -1e30f;
  for (int s = 0; s < kNS; ++s) mx = fmaxf(mx, (float)Y[base + (size_t)s * kOut]);
  outp[((size_t)b * kOut + o) * kNP + p] = mx;
}

__global__ void finalize_kernel(const float* __restrict__ newxyz,
                                const int* __restrict__ inds,
                                float* __restrict__ dout) {
  int i = blockIdx.x * 256 + threadIdx.x;
  if (i < kB * kNP * 3) dout[i] = newxyz[i];
  if (i < kB * kNP)
    reinterpret_cast<int*>(dout)[kB * kNP * 3 + (size_t)kB * kOut * kNP + i] = inds[i];
}

// ---------------------------------------------------------------------------
// host
// ---------------------------------------------------------------------------
extern "C" void kernel_launch(void* const* d_in, const int* in_sizes, int n_in,
                              void* d_out, int out_size, void* d_ws, size_t ws_size,
                              hipStream_t stream) {
  (void)in_sizes; (void)n_in; (void)out_size; (void)ws_size;
  int t = 0;
  auto F = [&](int i) { return (const float*)d_in[i]; };
  // jax pytree dict flatten = sorted keys: features, params{fc1,fc2,pe1,pe2,t1,t2}, xyz
  const float* FEAT = F(t++);
  const float* FC1_BN_B = F(t++); const float* FC1_BN_G = F(t++); const float* FC1_W = F(t++);
  const float* FC2_BN_B = F(t++); const float* FC2_BN_G = F(t++); const float* FC2_W = F(t++);
  const float* PE1_B2 = F(t++); const float* PE1_BN_B = F(t++); const float* PE1_BN_G = F(t++);
  const float* PE1_W1 = F(t++); const float* PE1_W2 = F(t++);
  const float* PE2_B2 = F(t++); const float* PE2_BN_B = F(t++); const float* PE2_BN_G = F(t++);
  const float* PE2_W1 = F(t++); const float* PE2_W2 = F(t++);
  const float* TP1_FF1_B = F(t++); const float* TP1_FF1_W = F(t++);
  const float* TP1_FF2_B = F(t++); const float* TP1_FF2_W = F(t++);
  const float* TP1_IN_B = F(t++);  const float* TP1_IN_W = F(t++);
  const float* TP1_LN1_B = F(t++); const float* TP1_LN1_G = F(t++);
  const float* TP1_LN2_B = F(t++); const float* TP1_LN2_G = F(t++);
  const float* TP1_OUT_B = F(t++); const float* TP1_OUT_W = F(t++);
  const float* TP2_FF1_B = F(t++); const float* TP2_FF1_W = F(t++);
  const float* TP2_FF2_B = F(t++); const float* TP2_FF2_W = F(t++);
  const float* TP2_IN_B = F(t++);  const float* TP2_IN_W = F(t++);
  const float* TP2_LN1_B = F(t++); const float* TP2_LN1_G = F(t++);
  const float* TP2_LN2_B = F(t++); const float* TP2_LN2_G = F(t++);
  const float* TP2_OUT_B = F(t++); const float* TP2_OUT_W = F(t++);
  const float* XYZ = F(t++);

  // workspace carve
  char* wsb = (char*)d_ws;
  size_t off = 0;
  auto carve = [&](size_t bytes) -> char* {
    char* p = wsb + off;
    off += bytes;
    off = (off + 255) & ~(size_t)255;
    return p;
  };
  int*   wInds  = (int*)  carve((size_t)kB * kNP * 4);
  float* wNew   = (float*)carve((size_t)kB * kNP * 3 * 4);
  int*   wKnn   = (int*)  carve((size_t)kM * 4);
  bf16*  wX     = (bf16*) carve((size_t)kM * kC * 2);
  bf16*  wT0    = (bf16*) carve((size_t)kM * 384 * 2);
  bf16*  wTa    = (bf16*) carve((size_t)kM * kC * 2);
  bf16*  wTb    = (bf16*) carve((size_t)kM * kC * 2);
  float* wCross = (float*)carve((size_t)kB * kC * kN * 4);
  float* wCnt   = (float*)carve((size_t)kB * kN * 4);
  bf16* wbIn1  = (bf16*)carve(384 * 128 * 2);
  bf16* wbOut1 = (bf16*)carve(128 * 128 * 2);
  bf16* wbF11  = (bf16*)carve(256 * 128 * 2);
  bf16* wbF21  = (bf16*)carve(128 * 256 * 2);
  bf16* wbIn2  = (bf16*)carve(384 * 128 * 2);
  bf16* wbOut2 = (bf16*)carve(128 * 128 * 2);
  bf16* wbF12  = (bf16*)carve(256 * 128 * 2);
  bf16* wbF22  = (bf16*)carve(128 * 256 * 2);
  bf16* wbFc1  = (bf16*)carve(128 * 128 * 2);
  bf16* wbFc2  = (bf16*)carve(256 * 128 * 2);

  auto cvt = [&](const float* s, bf16* d, int n) {
    f2bf_kernel<<<(n + 255) / 256, 256, 0, stream>>>(s, d, n);
  };
  cvt(TP1_IN_W, wbIn1, 384 * 128);  cvt(TP1_OUT_W, wbOut1, 128 * 128);
  cvt(TP1_FF1_W, wbF11, 256 * 128); cvt(TP1_FF2_W, wbF21, 128 * 256);
  cvt(TP2_IN_W, wbIn2, 384 * 128);  cvt(TP2_OUT_W, wbOut2, 128 * 128);
  cvt(TP2_FF1_W, wbF12, 256 * 128); cvt(TP2_FF2_W, wbF22, 128 * 256);
  cvt(FC1_W, wbFc1, 128 * 128);     cvt(FC2_W, wbFc2, 256 * 128);

  auto gemm = [&](const bf16* A, const bf16* Wt, const float* scale, float smul,
                  const float* bias, bf16* out, int M, int Nn, int K, int act) {
    dim3 g(Nn / 128, M / 128);
    gemm_bf16_kernel<<<g, 256, 0, stream>>>(A, Wt, scale, smul, bias, out, M, Nn, K, act);
  };

  auto encoder = [&](const bf16* inw, const float* inb, const bf16* outw,
                     const float* outb_, const float* ln1g, const float* ln1b,
                     const bf16* f1w, const float* f1b, const bf16* f2w,
                     const float* f2b, const float* ln2g, const float* ln2b) {
    gemm(wX, inw, nullptr, 1.0f, inb, wT0, kM, 384, 128, 0);          // qkv
    attn_kernel<<<(kB * kNP * 4) / 8, 256, 0, stream>>>(wT0, wTa);    // attention
    gemm(wTa, outw, nullptr, 1.0f, outb_, wTb, kM, 128, 128, 0);      // out proj
    add_ln_kernel<<<kM / 8, 256, 0, stream>>>(wX, wTb, ln1g, ln1b);   // ln1
    gemm(wX, f1w, nullptr, 1.0f, f1b, wT0, kM, 256, 128, 1);          // ff1+relu
    gemm(wT0, f2w, nullptr, 1.0f, f2b, wTb, kM, 128, 256, 0);         // ff2
    add_ln_kernel<<<kM / 8, 256, 0, stream>>>(wX, wTb, ln2g, ln2b);   // ln2
  };

  const float bnmul = 1.0f / sqrtf(1.0f + 1e-5f);

  fps_kernel<<<kB, 1024, 0, stream>>>(XYZ, wInds, wNew);
  knn_kernel<<<(kB * kNP) / 256, 256, 0, stream>>>(XYZ, wNew, wKnn);

  build_tokens_kernel<<<kB * kNP, 128, 0, stream>>>(
      FEAT, XYZ, wNew, wKnn, PE1_W1, PE1_BN_G, PE1_BN_B, PE1_W2, PE1_B2, wX);
  encoder(wbIn1, TP1_IN_B, wbOut1, TP1_OUT_B, TP1_LN1_G, TP1_LN1_B,
          wbF11, TP1_FF1_B, wbF21, TP1_FF2_B, TP1_LN2_G, TP1_LN2_B);
  gemm(wX, wbFc1, FC1_BN_G, bnmul, FC1_BN_B, wTa, kM, 128, 128, 1);   // fc1+bn+relu

  (void)hipMemsetAsync(wCross, 0, (size_t)kB * kC * kN * 4, stream);
  (void)hipMemsetAsync(wCnt, 0, (size_t)kB * kN * 4, stream);
  scatter_kernel<<<kM, 128, 0, stream>>>(wTa, wKnn, wCross, wCnt);
  divide_kernel<<<((size_t)kB * kC * kN + 255) / 256, 256, 0, stream>>>(wCross, wCnt);

  build_tokens_kernel<<<kB * kNP, 128, 0, stream>>>(
      wCross, XYZ, wNew, wKnn, PE2_W1, PE2_BN_G, PE2_BN_B, PE2_W2, PE2_B2, wX);
  encoder(wbIn2, TP2_IN_B, wbOut2, TP2_OUT_B, TP2_LN1_G, TP2_LN1_B,
          wbF12, TP2_FF1_B, wbF22, TP2_FF2_B, TP2_LN2_G, TP2_LN2_B);
  gemm(wX, wbFc2, FC2_BN_G, bnmul, FC2_BN_B, wT0, kM, 256, 128, 1);   // fc2+bn+relu

  maxpool_kernel<<<((size_t)kB * kOut * kNP + 255) / 256, 256, 0, stream>>>(
      wT0, (float*)d_out + (size_t)kB * kNP * 3);
  finalize_kernel<<<(kB * kNP * 3 + 255) / 256, 256, 0, stream>>>(
      wNew, wInds, (float*)d_out);
}
